// DyDCNv2_25374666785291
// MI455X (gfx1250) — matrix operand order repaired and checked by reading, not compiled, not run on hardware
//
#include <hip/hip_runtime.h>
#include <hip/hip_bf16.h>
#include <stdint.h>

// ---------------------------------------------------------------------------
// DCNv2 (3x3 deformable conv, modulated) + GroupNorm for MI455X (gfx1250).
// Shapes: B=4, C=256, Co=256, H=W=64, K=9, GROUPS=16.
// Pipeline:
//   0) zero stats
//   1) NCHW -> NHWC transpose of input (channel-contiguous bilinear gathers)
//   2) weight (Co,C,3,3) f32 -> bf16 reordered [o][k*256+c]
//   3) im2col bilinear sampling -> cols bf16 laid out [b][pixel][k*256+c]
//      (K-contiguous: exactly the layout both WMMA operands want in LDS)
//   4) tiled WMMA GEMM (M=Co=256, N=4096/batch, K=2304) with fused bias and
//      per-(batch,group) sum/sumsq reduction
//   5) GroupNorm finalize
// ---------------------------------------------------------------------------

#define BN    4
#define CCH   256
#define COUT  256
#define HH    64
#define WW    64
#define HWSZ  4096      // H*W
#define KPTS  9
#define KC    2304      // C*K
#define NGRP  16
#define GSIZE 16        // Co / NGRP

typedef __attribute__((ext_vector_type(16))) __bf16          v16bf;
typedef __attribute__((ext_vector_type(8)))  float           v8f;
typedef __attribute__((ext_vector_type(8)))  unsigned short  ushort8;

struct U16x16 { ushort8 lo; ushort8 hi; };

static __device__ __forceinline__ unsigned short f2bf(float f) {
  unsigned int u = __float_as_uint(f);
  u += 0x7FFFu + ((u >> 16) & 1u);          // round to nearest even
  return (unsigned short)(u >> 16);
}

static __device__ __forceinline__ v16bf make_frag(ushort8 lo, ushort8 hi) {
  U16x16 t{lo, hi};
  return __builtin_bit_cast(v16bf, t);
}

// --------------------------- kernel 0: zero stats ---------------------------
__global__ void zero_stats_kernel(float* stats) {
  stats[threadIdx.x] = 0.0f;                // 128 floats: [B][16][2]
}

// ----------------------- kernel 1: NCHW -> NHWC (f32) -----------------------
__global__ void __launch_bounds__(256) nchw_to_nhwc_kernel(
    const float* __restrict__ in, float* __restrict__ nhwc) {
  unsigned int idx = blockIdx.x * 256u + threadIdx.x;   // 4*4096*256 = 2^22
  int c  = idx & 255;
  int hw = (idx >> 8) & 4095;
  int b  = idx >> 20;
  nhwc[((size_t)b * HWSZ + hw) * CCH + c] =
      in[((size_t)b * CCH + c) * HWSZ + hw];
}

// --------------- kernel 2: weight f32 (Co,C,3,3) -> bf16 [o][k*256+c] -------
__global__ void __launch_bounds__(256) weight_to_bf16_kernel(
    const float* __restrict__ w, unsigned short* __restrict__ wbf) {
  unsigned int idx = blockIdx.x * 256u + threadIdx.x;   // 256*2304
  int rr = idx % KC;            // rr = k*256 + c
  int o  = idx / KC;
  int k  = rr >> 8;
  int c  = rr & 255;
  wbf[(size_t)o * KC + rr] = f2bf(w[((size_t)o * CCH + c) * KPTS + k]);
}

// ------------------- kernel 3: im2col with bilinear sampling ----------------
// One 32-lane group per sample point (b,k,hw); lanes sweep channels so the
// 4 corner gathers from NHWC are fully coalesced (128B) and the bf16 writes
// to cols[b][hw][k*256+c] are contiguous.
__global__ void __launch_bounds__(256) im2col_kernel(
    const float* __restrict__ nhwc,
    const float* __restrict__ offset,   // (B,18,H,W)
    const float* __restrict__ mask,     // (B,9,H,W)
    unsigned short* __restrict__ cols)  // [B][4096][2304] bf16
{
  unsigned int gid  = blockIdx.x * 256u + threadIdx.x;
  int lane = threadIdx.x & 31;
  unsigned int sp = gid >> 5;                  // sample point 0..147455
  int hw  = sp & 4095;
  unsigned int t = sp >> 12;
  int k = t % KPTS;
  int b = t / KPTS;
  int h = hw >> 6, w = hw & 63;

  float dy = offset[(((size_t)b * 18 + 2 * k    ) << 12) + hw];
  float dx = offset[(((size_t)b * 18 + 2 * k + 1) << 12) + hw];
  float mk = mask  [(((size_t)b * 9  + k        ) << 12) + hw];

  float y = (float)(h + (k / 3) - 1) + dy;
  float x = (float)(w + (k % 3) - 1) + dx;
  float y0 = floorf(y), x0 = floorf(x);
  float wy1 = y - y0, wx1 = x - x0;
  float wy0 = 1.0f - wy1, wx0 = 1.0f - wx1;

  float w00 = wy0 * wx0 * mk, w01 = wy0 * wx1 * mk;
  float w10 = wy1 * wx0 * mk, w11 = wy1 * wx1 * mk;

  // corner validity + clamped linear index
  float yc0 = y0, yc1 = y0 + 1.0f, xc0 = x0, xc1 = x0 + 1.0f;
  if (!(yc0 >= 0.0f && yc0 < (float)HH)) { w00 = 0.0f; w01 = 0.0f; }
  if (!(yc1 >= 0.0f && yc1 < (float)HH)) { w10 = 0.0f; w11 = 0.0f; }
  if (!(xc0 >= 0.0f && xc0 < (float)WW)) { w00 = 0.0f; w10 = 0.0f; }
  if (!(xc1 >= 0.0f && xc1 < (float)WW)) { w01 = 0.0f; w11 = 0.0f; }
  int yi0 = min(max((int)yc0, 0), HH - 1);
  int yi1 = min(max((int)yc1, 0), HH - 1);
  int xi0 = min(max((int)xc0, 0), WW - 1);
  int xi1 = min(max((int)xc1, 0), WW - 1);
  int i00 = yi0 * WW + xi0, i01 = yi0 * WW + xi1;
  int i10 = yi1 * WW + xi0, i11 = yi1 * WW + xi1;

  const float* base = nhwc + (size_t)b * HWSZ * CCH;
  unsigned short* dst = cols + ((size_t)b * HWSZ + hw) * KC + k * CCH;

  #pragma unroll
  for (int ci = 0; ci < CCH / 32; ++ci) {
    int c = ci * 32 + lane;
    float v = base[(size_t)i00 * CCH + c] * w00
            + base[(size_t)i01 * CCH + c] * w01
            + base[(size_t)i10 * CCH + c] * w10
            + base[(size_t)i11 * CCH + c] * w11;
    dst[c] = f2bf(v);
  }
}

// ----------------- kernel 4: WMMA GEMM + bias + group stats -----------------
// Per batch: gout[o][n] = sum_r wbf[o][r] * cols[n][r], r in [0,2304).
// Block: 256 threads (8 waves, wave32). Tile 128(M) x 128(N), K-step 32.
// Wave grid 2x4: each wave owns 4 M-subtiles x 2 N-subtiles of 16x16.
__global__ void __launch_bounds__(256) dcn_gemm_wmma_kernel(
    const unsigned short* __restrict__ wbf,   // [256][2304] bf16
    const unsigned short* __restrict__ cols,  // [B][4096][2304] bf16
    const float* __restrict__ bias,
    float* __restrict__ gout,                 // [B][256][4096] f32
    float* __restrict__ stats)                // [B][16][2]  (sum, sumsq)
{
  const int tid    = threadIdx.x;
  const int lane   = tid & 31;
  const int wave   = tid >> 5;
  const int waveM  = wave >> 2;   // 0..1
  const int waveN  = wave & 3;    // 0..3
  const int lane16 = lane & 15;
  const int laneHi = lane >> 4;

  const int nBase = blockIdx.x * 128;
  const int oBase = blockIdx.y * 128;
  const int b     = blockIdx.z;

  // [128 rows][32 k] bf16 tiles, padded to 40 (80B row stride, 16B aligned)
  __shared__ __attribute__((aligned(16))) unsigned short As[128 * 40];
  __shared__ __attribute__((aligned(16))) unsigned short Bs[128 * 40];
  __shared__ float sSum[8], sSsq[8];
  if (tid < 8) { sSum[tid] = 0.0f; sSsq[tid] = 0.0f; }

  const v8f vzero = {0.f, 0.f, 0.f, 0.f, 0.f, 0.f, 0.f, 0.f};
  v8f acc[4][2];
  #pragma unroll
  for (int i = 0; i < 4; ++i) { acc[i][0] = vzero; acc[i][1] = vzero; }

  const int row  = tid >> 1;           // 0..127
  const int colh = (tid & 1) << 4;     // 0 or 16 (bf16 elements)
  const unsigned short* gA = wbf  + (size_t)(oBase + row) * KC + colh;
  const unsigned short* gB = cols + ((size_t)b * HWSZ + (nBase + row)) * KC + colh;

  for (int kk = 0; kk < KC; kk += 32) {
    // ---- global -> LDS (each thread: 32B of A tile + 32B of B tile) ----
    const ushort8* pa = (const ushort8*)(gA + kk);
    const ushort8* pb = (const ushort8*)(gB + kk);
    *(ushort8*)&As[row * 40 + colh    ] = pa[0];
    *(ushort8*)&As[row * 40 + colh + 8] = pa[1];
    *(ushort8*)&Bs[row * 40 + colh    ] = pb[0];
    *(ushort8*)&Bs[row * 40 + colh + 8] = pb[1];
    if (kk + 32 < KC) {                       // global_prefetch_b8 next slab
      __builtin_prefetch(gA + kk + 32, 0, 1);
      __builtin_prefetch(gB + kk + 32, 0, 1);
    }
    __syncthreads();

    // ---- build WMMA fragments from LDS (ds_load_b128 pairs) ----
    // A (16x32 bf16): lanes 0-15 hold k in {0..7}u{16..23}, lanes 16-31
    // hold k in {8..15}u{24..31}; m = lane%16.
    const int kbA = laneHi ? 8 : 0;
    // B (32x16 bf16): lanes 0-15 hold k in {0..7}u{16..23}, lanes 16-31
    // hold k in {8..15}u{24..31}; n=lane%16.
    const int kbB = laneHi ? 8 : 0;

    v16bf afr[4], bfr[2];
    #pragma unroll
    for (int sm = 0; sm < 4; ++sm) {
      const unsigned short* p = &As[(waveM * 64 + sm * 16 + lane16) * 40 + kbA];
      afr[sm] = make_frag(*(const ushort8*)p, *(const ushort8*)(p + 16));
    }
    #pragma unroll
    for (int sn = 0; sn < 2; ++sn) {
      const unsigned short* p = &Bs[(waveN * 32 + sn * 16 + lane16) * 40 + kbB];
      bfr[sn] = make_frag(*(const ushort8*)p, *(const ushort8*)(p + 16));
    }

    #pragma unroll
    for (int sm = 0; sm < 4; ++sm)
      #pragma unroll
      for (int sn = 0; sn < 2; ++sn)
        acc[sm][sn] = __builtin_amdgcn_wmma_f32_16x16x32_bf16(
            false, afr[sm], false, bfr[sn], (short)0, acc[sm][sn],
            false, false);
    __syncthreads();
  }

  // ---- epilogue: add bias, store f32, accumulate per-group sum/sumsq ----
  // D layout (16x16 f32): VGPR r, lane L -> row r + 8*(L/16), col L%16.
  // Each 16-row M-subtile is exactly one GN group (GSIZE==16).
  #pragma unroll
  for (int sm = 0; sm < 4; ++sm) {
    float s = 0.0f, ss = 0.0f;
    const int mBase = oBase + waveM * 64 + sm * 16 + laneHi * 8;
    #pragma unroll
    for (int sn = 0; sn < 2; ++sn) {
      const int n = nBase + waveN * 32 + sn * 16 + lane16;
      float* orow = gout + ((size_t)b * COUT + mBase) * HWSZ + n;
      #pragma unroll
      for (int r = 0; r < 8; ++r) {
        float v = acc[sm][sn][r] + bias[mBase + r];
        orow[(size_t)r * HWSZ] = v;
        s += v; ss += v * v;
      }
    }
    atomicAdd(&sSum[waveM * 4 + sm], s);     // ds_add_f32
    atomicAdd(&sSsq[waveM * 4 + sm], ss);
  }
  __syncthreads();
  if (tid < 8) {
    const int g = (oBase >> 4) + tid;        // global group 0..15
    atomicAdd(&stats[((size_t)b * NGRP + g) * 2 + 0], sSum[tid]);
    atomicAdd(&stats[((size_t)b * NGRP + g) * 2 + 1], sSsq[tid]);
  }
}

// ------------------------ kernel 5: GroupNorm finalize ----------------------
__global__ void __launch_bounds__(256) gn_finalize_kernel(
    const float* __restrict__ gemm, const float* __restrict__ stats,
    const float* __restrict__ gamma, const float* __restrict__ beta,
    float* __restrict__ out)
{
  unsigned int idx = blockIdx.x * 256u + threadIdx.x;   // 2^22
  int o = (idx >> 12) & 255;
  int b = idx >> 20;
  int g = o >> 4;
  float s  = stats[((size_t)b * NGRP + g) * 2 + 0];
  float ss = stats[((size_t)b * NGRP + g) * 2 + 1];
  const float inv = 1.0f / (float)(GSIZE * HWSZ);       // 1/65536
  float mu  = s * inv;
  float var = ss * inv - mu * mu;
  float rstd = rsqrtf(var + 1e-5f);
  float v = gemm[idx];
  out[idx] = (v - mu) * rstd * gamma[o] + beta[o];
}

// ---------------------------------------------------------------------------
extern "C" void kernel_launch(void* const* d_in, const int* in_sizes, int n_in,
                              void* d_out, int out_size, void* d_ws, size_t ws_size,
                              hipStream_t stream) {
  const float* input  = (const float*)d_in[0];  // (4,256,64,64)
  const float* offset = (const float*)d_in[1];  // (4,18,64,64)
  const float* maskp  = (const float*)d_in[2];  // (4,9,64,64)
  const float* weight = (const float*)d_in[3];  // (256,256,3,3)
  const float* biasp  = (const float*)d_in[4];  // (256,)
  const float* gammap = (const float*)d_in[5];  // (256,)
  const float* betap  = (const float*)d_in[6];  // (256,)

  char* ws = (char*)d_ws;
  float*          nhwc  = (float*)(ws);                         // 16 MB
  unsigned short* wbf   = (unsigned short*)(ws + 16777216);     // 1.125 MB
  unsigned short* cols  = (unsigned short*)(ws + 17956864);     // 72 MB
  float*          gout  = (float*)(ws + 93454336);              // 16 MB
  float*          stats = (float*)(ws + 110231552);             // 512 B

  zero_stats_kernel<<<1, 128, 0, stream>>>(stats);
  nchw_to_nhwc_kernel<<<16384, 256, 0, stream>>>(input, nhwc);
  weight_to_bf16_kernel<<<2304, 256, 0, stream>>>(weight, wbf);
  im2col_kernel<<<18432, 256, 0, stream>>>(nhwc, offset, maskp, cols);
  dim3 ggrid(HWSZ / 128, COUT / 128, BN);                       // 32 x 2 x 4
  dcn_gemm_wmma_kernel<<<ggrid, 256, 0, stream>>>(wbf, cols, biasp, gout, stats);
  gn_finalize_kernel<<<16384, 256, 0, stream>>>(gout, stats, gammap, betap,
                                                (float*)d_out);
}
